// LIFNeuronFDE_17738214932777
// MI455X (gfx1250) — matrix operand order repaired and checked by reading, not compiled, not run on hardware
//
#include <hip/hip_runtime.h>

typedef __attribute__((ext_vector_type(16))) __bf16 v16bf;
typedef __attribute__((ext_vector_type(8)))  float  v8f;

#define TSTEPS 32
#define BLOCK  256
#define WAVES  (BLOCK / 32)

// GL coefficients for beta = 0.5:  c_0 = 1, c_j = (1 - 1.5/j) * c_{j-1}
static constexpr float glcf(int j) {
  double c = 1.0;
  for (int i = 1; i <= j; ++i) c *= (1.0 - 1.5 / (double)i);
  return (float)c;
}

__device__ __constant__ float GLC[TSTEPS + 1] = {
  glcf(0),  glcf(1),  glcf(2),  glcf(3),  glcf(4),  glcf(5),  glcf(6),  glcf(7),
  glcf(8),  glcf(9),  glcf(10), glcf(11), glcf(12), glcf(13), glcf(14), glcf(15),
  glcf(16), glcf(17), glcf(18), glcf(19), glcf(20), glcf(21), glcf(22), glcf(23),
  glcf(24), glcf(25), glcf(26), glcf(27), glcf(28), glcf(29), glcf(30), glcf(31),
  glcf(32)
};

__device__ __forceinline__ float fast_sigmoid5(float x) {
  // sigmoid(5*(x-1)):  exp(-5(x-1)) = exp2(-5*log2(e)*(x-1)); single exp + single rcp
  float e = __builtin_amdgcn_exp2f(-7.2134752044448169f * (x - 1.0f));
  return __builtin_amdgcn_rcpf(1.0f + e);
}

__global__ __launch_bounds__(BLOCK)
void lif_fde_kernel(const float* __restrict__ I,
                    const float* __restrict__ v0,
                    float* __restrict__ out,
                    int BN) {
  // Per-wave staging tile: [wave][time/step][neuron(0..31)+pad]
  __shared__ float stage[WAVES][16][33];

  const int tid  = blockIdx.x * BLOCK + threadIdx.x;   // global neuron id
  const int lane = threadIdx.x & 31;
  const int w    = threadIdx.x >> 5;
  const int m    = lane & 15;
  const bool hi  = lane >= 16;

  const float hb_tau = 2.0f;  // h^beta / tau = 1.0 / 0.5

  float v = v0[tid];

  // Batch-issue phase-A input loads (16 independent NT loads in flight per wave).
  float Ia[16];
  #pragma unroll
  for (int k = 0; k < 16; ++k)
    Ia[k] = __builtin_nontemporal_load(&I[(size_t)k * BN + tid]);

  // ---- Phase A: steps 0..15, full triangular GL sum in registers ----
  float h0[16];
  #pragma unroll
  for (int k = 0; k < 16; ++k) {
    h0[k] = v;
    float mem = 0.0f;
    #pragma unroll
    for (int j = 0; j <= k; ++j)
      mem = fmaf(GLC[k + 1 - j], h0[j], mem);
    v = hb_tau * (Ia[k] - v) - mem;
    __builtin_nontemporal_store(fast_sigmoid5(v), &out[(size_t)k * BN + tid]);
  }

  // Batch-issue phase-C input loads now so they overlap LDS staging + WMMA.
  float Ic[16];
  #pragma unroll
  for (int k = 0; k < 16; ++k)
    Ic[k] = __builtin_nontemporal_load(&I[(size_t)(k + 16) * BN + tid]);

  // Stage history block 0..15 into this wave's LDS tile (neuron = lane).
  #pragma unroll
  for (int j = 0; j < 16; ++j) stage[w][j][lane] = h0[j];
  __syncthreads();

  // ---- Cross-block GL term via WMMA ----
  // A (16x32 bf16): A[m][j] = c_{(16+m)+1-j} for j = 0..15, zero for j = 16..31.
  // A layout (contiguous, matching B): lanes 0-15 hold K=0..15 (element e = K); lanes 16-31 hold K=16..31.
  v16bf a;
  #pragma unroll
  for (int e = 0; e < 16; ++e) {
    a[e] = hi ? (__bf16)0.0f               // K = 16..31 columns are zero
              : (__bf16)GLC[17 + m - e];   // K = e -> c[17+m-e]
  }

  // B (32x16 bf16): rows K = history index j, cols N = neurons.
  // Layout: lanes 0-15 hold K=0..15 (element e = K); lanes 16-31 hold K=16..31 (zero here).
  v16bf b0, b1;
  #pragma unroll
  for (int e = 0; e < 16; ++e) {
    float x0 = hi ? 0.0f : stage[w][e][m];        // neurons 0..15 of this wave
    float x1 = hi ? 0.0f : stage[w][e][m + 16];   // neurons 16..31
    b0[e] = (__bf16)x0;
    b1[e] = (__bf16)x1;
  }
  __syncthreads();

  v8f d0 = {0.f,0.f,0.f,0.f,0.f,0.f,0.f,0.f};
  v8f d1 = {0.f,0.f,0.f,0.f,0.f,0.f,0.f,0.f};
  d0 = __builtin_amdgcn_wmma_f32_16x16x32_bf16(false, a, false, b0,
                                               (short)0, d0, false, false);
  d1 = __builtin_amdgcn_wmma_f32_16x16x32_bf16(false, a, false, b1,
                                               (short)0, d1, false, false);

  // D layout (f32 16x16): lanes 0-15 vgpr r -> (M=r, N=lane); lanes 16-31 -> (M=r+8, N=lane-16).
  // Scatter to stage[w][step-16][neuron].
  #pragma unroll
  for (int r = 0; r < 8; ++r) {
    int mm = hi ? (r + 8) : r;
    stage[w][mm][m]      = d0[r];
    stage[w][mm][m + 16] = d1[r];
  }
  __syncthreads();

  // ---- Phase C: steps 16..31 = WMMA cross term + in-block triangular sum ----
  float h1[16];
  #pragma unroll
  for (int k = 16; k < 32; ++k) {
    const int kk = k - 16;
    h1[kk] = v;
    float mem = stage[w][kk][lane];               // Σ_{j=0..15} c_{k+1-j} v_j (WMMA)
    #pragma unroll
    for (int j = 16; j <= k; ++j)
      mem = fmaf(GLC[k + 1 - j], h1[j - 16], mem);
    v = hb_tau * (Ic[kk] - v) - mem;
    __builtin_nontemporal_store(fast_sigmoid5(v), &out[(size_t)k * BN + tid]);
  }
}

extern "C" void kernel_launch(void* const* d_in, const int* in_sizes, int n_in,
                              void* d_out, int out_size, void* d_ws, size_t ws_size,
                              hipStream_t stream) {
  const float* I  = (const float*)d_in[0];   // (T, B, N) float32
  const float* v0 = (const float*)d_in[1];   // (B, N)    float32
  float* out      = (float*)d_out;           // (T, B, N) float32
  (void)n_in; (void)out_size; (void)d_ws; (void)ws_size;

  const int BN = in_sizes[1];                // B*N = 1,048,576 (exact multiple of BLOCK)
  dim3 grid(BN / BLOCK), block(BLOCK);
  lif_fde_kernel<<<grid, block, 0, stream>>>(I, v0, out, BN);
}
